// RayBasedGrouping_23922967838987
// MI455X (gfx1250) — matrix operand order repaired and checked by reading, not compiled, not run on hardware
//
#include <hip/hip_runtime.h>
#include <cmath>

#ifndef M_PI
#define M_PI 3.14159265358979323846
#endif

// ---------------- problem constants ----------------
static constexpr int kB     = 2;
static constexpr int kP     = 128;
static constexpr int kRAY   = 66;
static constexpr int kBIN   = 4;
static constexpr int kFBIN  = 2;
static constexpr int kSEED  = 256;
static constexpr int kRED   = 64;
static constexpr int kHALF  = 32;
static constexpr int kAGG   = 128;
static constexpr int kNPTS  = 16384;
static constexpr int kFPSM  = 1024;
static constexpr int kNSEED = 1024;
static constexpr int kNS    = 16;        // SA_NS == FSA_NS
static constexpr int kNQC   = kP * kBIN * kRAY;   // 33792
static constexpr int kNQF   = kP * kFBIN * kRAY;  // 16896
static constexpr float kSAR2  = 0.4f * 0.4f;
static constexpr float kFSAR2 = 0.2f * 0.2f;

typedef __attribute__((ext_vector_type(16))) _Float16 v16h;
typedef __attribute__((ext_vector_type(8)))  float    v8f;
typedef __attribute__((ext_vector_type(4)))  unsigned int tdm_g0_t;
typedef __attribute__((ext_vector_type(8)))  int          tdm_g1_t;
typedef __attribute__((ext_vector_type(4)))  int          tdm_g23_t;
typedef __attribute__((ext_vector_type(8)))  int          tdm_g4_t;

// ---- Tensor Data Mover: DMA a 32x16 f32 tile of a row-major [*,N] matrix to LDS ----
// D# built per CDNA5 ISA 8.3/8.4: group0 {count=1, lds_addr, global_addr, type=2},
// group1 {data_size=4B, tensor_dim0=N, tensor_dim1=32, tile_dim0=16, tile_dim1=32,
// tensor_dim0_stride=N}. Groups 2/3 unused (2-D tensor). Tracked by TENSORcnt.
// This toolchain exposes the 6-arg builtin (4 descriptor groups + extra group + cpol).
__device__ __forceinline__ void tdm_load_b32x16(const float* gsrc, unsigned lds_off,
                                                unsigned N) {
  unsigned long long ga = (unsigned long long)(const void*)gsrc;
  tdm_g0_t g0;
  g0[0] = 1u;                                                 // count=1 (valid, user)
  g0[1] = lds_off;                                            // LDS byte address
  g0[2] = (unsigned)ga;                                       // global_addr[31:0]
  g0[3] = (unsigned)((ga >> 32) & 0x01ffffffu) | (2u << 30);  // addr[56:32] | type=2
  tdm_g1_t g1;
  g1[0] = (int)(2u << 16);                                    // data_size = 4 bytes
  g1[1] = (int)((N & 0xffffu) << 16);                         // tensor_dim0 lo16
  g1[2] = (int)(((N >> 16) & 0xffffu) | (32u << 16));         // dim0 hi | tensor_dim1=32
  g1[3] = (int)(16u << 16);                                   // dim1 hi=0 | tile_dim0=16
  g1[4] = 32;                                                 // tile_dim1=32, tile_dim2=0
  g1[5] = (int)N;                                             // tensor_dim0_stride lo32
  g1[6] = 0;                                                  // stride hi | dim1_stride lo
  g1[7] = 0;
  tdm_g23_t gz = {0, 0, 0, 0};
  tdm_g4_t  g4 = {0, 0, 0, 0, 0, 0, 0, 0};
  __builtin_amdgcn_tensor_load_to_lds(g0, g1, gz, gz, g4, 0);
}

// ---------------- ray table (generate_ray on device) ----------------
__global__ void rays_kernel(float* __restrict__ rays) {
  if (threadIdx.x != 0 || blockIdx.x != 0) return;
  const int n = 4;                          // ceil(sqrt((66-2)/4))
  double az_step = 0.5 * M_PI / n;
  double az = 0.0;
  int o = 0;
  for (int a = -n; a <= n; ++a) {
    double pol = 0.0;
    int size = (n - (a < 0 ? -a : a)) * 4;
    if (size == 0) size = 1;
    double step = 2.0 * M_PI / size;
    for (int k = 0; k < size; ++k) {
      pol += step;
      double r = sin(az);
      rays[o * 3 + 0] = (float)(cos(pol) * r);
      rays[o * 3 + 1] = (float)(sin(pol) * r);
      rays[o * 3 + 2] = (float)cos(az);
      ++o;
    }
    az += az_step;
  }
}

// ---------------- FPS: entire point cloud staged in 320KB WGP LDS ----------------
// dynamic LDS layout (floats): xyz SoA 3*16384, redV 1024, redI 1024, fidx 1024
__global__ void __launch_bounds__(1024)
fps_kernel(const float* __restrict__ pts, float* __restrict__ tgt) {
  extern __shared__ float s[];
  float* sx   = s;
  float* sy   = s + kNPTS;
  float* sz   = s + 2 * kNPTS;
  float* redV = s + 3 * kNPTS;
  int*   redI = (int*)(s + 3 * kNPTS + 1024);
  int*   fidx = (int*)(s + 3 * kNPTS + 2048);
  const int bb = blockIdx.x, t = threadIdx.x;
  const float* p = pts + (long long)bb * kNPTS * 3;
  for (int i = t; i < kNPTS; i += 1024) {
    sx[i] = p[i * 3 + 0]; sy[i] = p[i * 3 + 1]; sz[i] = p[i * 3 + 2];
  }
  float dist[16], px[16], py[16], pz[16];
  #pragma unroll
  for (int i = 0; i < 16; ++i) {
    int pi = t * 16 + i;
    px[i] = p[pi * 3 + 0]; py[i] = p[pi * 3 + 1]; pz[i] = p[pi * 3 + 2];
    dist[i] = 1e10f;
  }
  if (t == 0) fidx[0] = 0;
  __syncthreads();
  int last = 0;
  for (int it = 1; it < kFPSM; ++it) {
    float lx = sx[last], ly = sy[last], lz = sz[last];
    float bv = -1.0f; int bi = t * 16;
    #pragma unroll
    for (int i = 0; i < 16; ++i) {
      float dx = px[i] - lx, dy = py[i] - ly, dz = pz[i] - lz;
      float d  = dx * dx + dy * dy + dz * dz;
      if (d < dist[i]) dist[i] = d;
      if (dist[i] > bv) { bv = dist[i]; bi = t * 16 + i; }
    }
    redV[t] = bv; redI[t] = bi;
    __syncthreads();
    for (int off = 512; off > 0; off >>= 1) {
      if (t < off) {
        float v2 = redV[t + off]; int i2 = redI[t + off];
        if (v2 > redV[t] || (v2 == redV[t] && i2 < redI[t])) { redV[t] = v2; redI[t] = i2; }
      }
      __syncthreads();
    }
    last = redI[0];
    if (t == 0) fidx[it] = last;
    __syncthreads();
  }
  if (t < kFPSM) {
    int id = fidx[t];
    float* o = tgt + ((long long)bb * kFPSM + t) * 3;
    o[0] = sx[id]; o[1] = sy[id]; o[2] = sz[id];
  }
}

// ---------------- three_nn + interpolation weights ----------------
__global__ void nn_kernel(const float* __restrict__ tgt, const float* __restrict__ seed_xyz,
                          int* __restrict__ nn_idx, float* __restrict__ nn_w) {
  int q = blockIdx.x * blockDim.x + threadIdx.x;
  int bb = blockIdx.z;
  if (q >= kFPSM) return;
  const float* tp = tgt + ((long long)bb * kFPSM + q) * 3;
  float qx = tp[0], qy = tp[1], qz = tp[2];
  const float* sp = seed_xyz + (long long)bb * kNSEED * 3;
  float d0 = 1e30f, d1 = 1e30f, d2 = 1e30f;
  int i0 = 0, i1 = 0, i2 = 0;
  for (int sI = 0; sI < kNSEED; ++sI) {
    float dx = qx - sp[sI * 3 + 0], dy = qy - sp[sI * 3 + 1], dz = qz - sp[sI * 3 + 2];
    float d  = dx * dx + dy * dy + dz * dz;
    if (d < d0)      { d2 = d1; i2 = i1; d1 = d0; i1 = i0; d0 = d; i0 = sI; }
    else if (d < d1) { d2 = d1; i2 = i1; d1 = d; i1 = sI; }
    else if (d < d2) { d2 = d; i2 = sI; }
  }
  float r0 = 1.0f / (d0 + 1e-8f), r1 = 1.0f / (d1 + 1e-8f), r2 = 1.0f / (d2 + 1e-8f);
  float inv = 1.0f / (r0 + r1 + r2);
  long long base = ((long long)bb * kFPSM + q) * 3;
  nn_idx[base + 0] = i0; nn_idx[base + 1] = i1; nn_idx[base + 2] = i2;
  nn_w[base + 0] = r0 * inv; nn_w[base + 1] = r1 * inv; nn_w[base + 2] = r2 * inv;
}

__global__ void interp_gather_kernel(const float* __restrict__ seed_feat,
                                     const int* __restrict__ nn_idx,
                                     const float* __restrict__ nn_w,
                                     float* __restrict__ out) {
  int t = blockIdx.x * blockDim.x + threadIdx.x;
  int bb = blockIdx.z;
  if (t >= kSEED * kFPSM) return;
  int q = t % kFPSM, c = t / kFPSM;
  long long nb = ((long long)bb * kFPSM + q) * 3;
  const float* f = seed_feat + (long long)bb * kSEED * kNSEED + (long long)c * kNSEED;
  float v = nn_w[nb + 0] * f[nn_idx[nb + 0]] +
            nn_w[nb + 1] * f[nn_idx[nb + 1]] +
            nn_w[nb + 2] * f[nn_idx[nb + 2]];
  out[(long long)bb * kSEED * kFPSM + (long long)c * kFPSM + q] = v;
}

// ---------------- generic WMMA GEMM: C = act(A[M,K] * B[K,N] + bias) ----------------
// A row-major (lda, shared over batch) streamed through L1 with global_prefetch;
// B panel staged 32x16 per k-step into LDS by the Tensor Data Mover, double-buffered,
// synchronized with s_wait_tensorcnt; compute via v_wmma_f32_16x16x32_f16.
// Requires K % 32 == 0, M % 16 == 0, N % 16 == 0.
__global__ void __launch_bounds__(32)
gemm16_wmma(const float* __restrict__ A, int lda,
            const float* __restrict__ Bm, long long sB,
            const float* __restrict__ bias,
            float* __restrict__ C, long long sC,
            int N, int K, int relu) {
  __shared__ float lbuf[2][32 * 16];
  const int lane = threadIdx.x;
  const int half = lane >> 4;
  const int l16  = lane & 15;
  const int n0 = blockIdx.x * 16;
  const int m0 = blockIdx.y * 16;
  const int bb = blockIdx.z;
  const float* Bp = Bm + (long long)bb * sB;
  float* Cp = C + (long long)bb * sC;
  const unsigned lds0 = (unsigned)(unsigned long long)(void*)&lbuf[0][0];
  const unsigned lds1 = (unsigned)(unsigned long long)(void*)&lbuf[1][0];
  // prologue: async-stage first B chunk
  tdm_load_b32x16(Bp + n0, lds0, (unsigned)N);
  v8f acc = {0.f, 0.f, 0.f, 0.f, 0.f, 0.f, 0.f, 0.f};
  for (int k0 = 0; k0 < K; k0 += 32) {
    const int cur = (k0 >> 5) & 1;
    if (k0 + 32 < K) {   // stage next chunk into the other buffer, wait for current
      tdm_load_b32x16(Bp + (long long)(k0 + 32) * N + n0, cur ? lds0 : lds1, (unsigned)N);
      __builtin_amdgcn_s_wait_tensorcnt(1);
    } else {
      __builtin_amdgcn_s_wait_tensorcnt(0);
    }
    v16h av, bv;
    const float* Ar = A + (long long)(m0 + l16) * lda + k0 + half * 8;
    __builtin_prefetch(Ar + 32, 0, 0);           // next A chunk -> global_prefetch_b8
    #pragma unroll
    for (int j = 0; j < 8; ++j) {
      av[j]     = (_Float16)Ar[j];
      av[j + 8] = (_Float16)Ar[16 + j];
    }
    const float* Lr = &lbuf[cur][half * 8 * 16 + l16];
    #pragma unroll
    for (int j = 0; j < 8; ++j) {
      bv[j]     = (_Float16)Lr[j * 16];
      bv[j + 8] = (_Float16)Lr[(16 + j) * 16];
    }
    acc = __builtin_amdgcn_wmma_f32_16x16x32_f16(false, av, false, bv,
                                                 (short)0, acc, false, false);
  }
  if (n0 + l16 < N) {
    #pragma unroll
    for (int r = 0; r < 8; ++r) {
      int m = m0 + half * 8 + r;
      float v = acc[r];
      if (bias) v += bias[m];
      if (relu && v < 0.f) v = 0.f;
      Cp[(long long)m * N + n0 + l16] = v;
    }
  }
}

// ---------------- query position generation ----------------
__global__ void cpos_kernel(const float* __restrict__ rays, const float* __restrict__ scale,
                            const float* __restrict__ ref, float* __restrict__ cpos) {
  int t = blockIdx.x * blockDim.x + threadIdx.x;
  int bb = blockIdx.z;
  if (t >= kP * kBIN * kRAY) return;
  int j = t % (kBIN * kRAY), p = t / (kBIN * kRAY);
  int bin = j / kRAY, ray = j % kRAY;
  float f = scale[bb * kP + p] * (float)(kBIN - bin) * (1.0f / kBIN);
  const float* rf = ref + ((long long)bb * kP + p) * 3;
  float* o = cpos + ((long long)bb * kNQC + t) * 3;
  o[0] = rays[ray * 3 + 0] * f + rf[0];
  o[1] = rays[ray * 3 + 1] * f + rf[1];
  o[2] = rays[ray * 3 + 2] * f + rf[2];
}

__global__ void dmax_kernel(const float* __restrict__ scale, const int* __restrict__ mask,
                            float* __restrict__ dmax) {
  int t = blockIdx.x * blockDim.x + threadIdx.x;
  int bb = blockIdx.z;
  if (t >= kP * kRAY) return;
  int ray = t % kRAY, p = t / kRAY;
  float sc = scale[bb * kP + p];
  float mx = -1e30f;
  #pragma unroll
  for (int bin = 0; bin < kBIN; ++bin) {
    float tv = sc * (float)(kBIN - bin) * (1.0f / kBIN);
    tv *= (float)mask[(long long)bb * kNQC + p * (kBIN * kRAY) + bin * kRAY + ray];
    if (tv > mx) mx = tv;
  }
  dmax[(long long)bb * kP * kRAY + t] = mx;
}

__global__ void fpos_kernel(const float* __restrict__ rays, const float* __restrict__ dmax,
                            const float* __restrict__ ref, float* __restrict__ fpos) {
  int t = blockIdx.x * blockDim.x + threadIdx.x;
  int bb = blockIdx.z;
  if (t >= kP * kFBIN * kRAY) return;
  int j = t % (kFBIN * kRAY), p = t / (kFBIN * kRAY);
  int fbin = j / kRAY, ray = j % kRAY;
  float f = dmax[(long long)bb * kP * kRAY + p * kRAY + ray] *
            (float)(kFBIN - fbin) * (1.0f / kFBIN);
  const float* rf = ref + ((long long)bb * kP + p) * 3;
  float* o = fpos + ((long long)bb * kNQF + t) * 3;
  o[0] = rays[ray * 3 + 0] * f + rf[0];
  o[1] = rays[ray * 3 + 1] * f + rf[1];
  o[2] = rays[ray * 3 + 2] * f + rf[2];
}

// ---------------- ball query: wave32 ballot + prefix-popcount compaction ----------------
__global__ void __launch_bounds__(32)
ballq_kernel(const float* __restrict__ tgt, const float* __restrict__ qpos,
             int* __restrict__ idxbuf, int* __restrict__ cntbuf, float r2, int NQ) {
  __shared__ int list[16];
  const int q = blockIdx.x, bb = blockIdx.z, lane = threadIdx.x;
  const float* qp = qpos + ((long long)bb * NQ + q) * 3;
  float qx = qp[0], qy = qp[1], qz = qp[2];
  const float* pts = tgt + (long long)bb * kFPSM * 3;
  int cnt = 0;
  for (int base = 0; base < kFPSM && cnt < kNS; base += 32) {
    int i = base + lane;
    float dx = pts[i * 3 + 0] - qx, dy = pts[i * 3 + 1] - qy, dz = pts[i * 3 + 2] - qz;
    bool in = (dx * dx + dy * dy + dz * dz) < r2;
    unsigned long long m = __ballot(in);
    int pre = (int)__popcll(m & ((1ull << lane) - 1ull));
    if (in) { int pos = cnt + pre; if (pos < kNS) list[pos] = i; }
    cnt += (int)__popcll(m);
    if (cnt > kNS) cnt = kNS;
  }
  __syncthreads();
  if (lane < kNS) {
    int v = 0;
    if (cnt > 0) v = (lane < cnt) ? list[lane] : list[0];
    idxbuf[((long long)bb * NQ + q) * kNS + lane] = v;
  }
  if (lane == 0) cntbuf[(long long)bb * NQ + q] = cnt;
}

// ---------------- SA module: gather + 32x64 WMMA conv + relu + max over samples ----------
__global__ void __launch_bounds__(32)
sa_wmma(const float* __restrict__ W, const float* __restrict__ bias,
        const float* __restrict__ feat, const int* __restrict__ idxbuf,
        const int* __restrict__ cntbuf, float* __restrict__ out, int NQ) {
  const int lane = threadIdx.x, half = lane >> 4, l16 = lane & 15;
  const int bb = blockIdx.z;
  v16h a00, a01, a10, a11;
  {
    const float* W0 = W + l16 * 64;
    const float* W1 = W + (16 + l16) * 64;
    #pragma unroll
    for (int j = 0; j < 8; ++j) {
      a00[j] = (_Float16)W0[half * 8 + j];        a00[j + 8] = (_Float16)W0[half * 8 + 16 + j];
      a01[j] = (_Float16)W0[32 + half * 8 + j];   a01[j + 8] = (_Float16)W0[32 + half * 8 + 16 + j];
      a10[j] = (_Float16)W1[half * 8 + j];        a10[j + 8] = (_Float16)W1[half * 8 + 16 + j];
      a11[j] = (_Float16)W1[32 + half * 8 + j];   a11[j + 8] = (_Float16)W1[32 + half * 8 + 16 + j];
    }
  }
  const float* fp = feat + (long long)bb * kRED * kFPSM;
  float* op = out + (long long)bb * kHALF * NQ;
  for (int qi = 0; qi < 8; ++qi) {
    int q = blockIdx.x * 8 + qi;
    int idn = idxbuf[((long long)bb * NQ + q) * kNS + l16];
    int cnt = cntbuf[(long long)bb * NQ + q];
    v16h b0, b1;
    #pragma unroll
    for (int j = 0; j < 16; ++j) {
      int c = (j < 8) ? half * 8 + j : 16 + half * 8 + (j - 8);
      b0[j] = (_Float16)fp[(long long)c * kFPSM + idn];
      b1[j] = (_Float16)fp[(long long)(32 + c) * kFPSM + idn];
    }
    v8f acc0 = {0.f,0.f,0.f,0.f,0.f,0.f,0.f,0.f};
    v8f acc1 = {0.f,0.f,0.f,0.f,0.f,0.f,0.f,0.f};
    acc0 = __builtin_amdgcn_wmma_f32_16x16x32_f16(false, a00, false, b0, (short)0, acc0, false, false);
    acc0 = __builtin_amdgcn_wmma_f32_16x16x32_f16(false, a01, false, b1, (short)0, acc0, false, false);
    acc1 = __builtin_amdgcn_wmma_f32_16x16x32_f16(false, a10, false, b0, (short)0, acc1, false, false);
    acc1 = __builtin_amdgcn_wmma_f32_16x16x32_f16(false, a11, false, b1, (short)0, acc1, false, false);
    float has = cnt > 0 ? 1.0f : 0.0f;
    #pragma unroll
    for (int r = 0; r < 8; ++r) {
      float v0 = acc0[r], v1 = acc1[r];
      #pragma unroll
      for (int off = 8; off >= 1; off >>= 1) {      // max over 16 samples, within half
        v0 = fmaxf(v0, __shfl_xor(v0, off));
        v1 = fmaxf(v1, __shfl_xor(v1, off));
      }
      if (l16 == 0) {
        int m0 = half * 8 + r, m1 = 16 + half * 8 + r;
        float o0 = v0 + bias[m0]; o0 = o0 > 0.f ? o0 : 0.f;
        float o1 = v1 + bias[m1]; o1 = o1 > 0.f ? o1 : 0.f;
        op[(long long)m0 * NQ + q] = o0 * has;
        op[(long long)m1 * NQ + q] = o1 * has;
      }
    }
  }
}

// ---------------- inter: combine + layer2 + argmax mask + score output ----------------
__global__ void inter2_kernel(const float* __restrict__ gb, const float* __restrict__ ha,
                              const float* __restrict__ b1, const float* __restrict__ W2,
                              const float* __restrict__ b2, float* __restrict__ score,
                              int* __restrict__ mask, int NQ, int JP) {
  int n = blockIdx.x * blockDim.x + threadIdx.x;
  int bb = blockIdx.z;
  if (n >= NQ) return;
  int p = n / JP;
  float h[32];
  #pragma unroll
  for (int c = 0; c < 32; ++c) {
    float v = gb[(long long)bb * 32 * NQ + (long long)c * NQ + n] +
              ha[(long long)bb * 32 * kP + c * kP + p] + b1[c];
    h[c] = v > 0.f ? v : 0.f;
  }
  float s0 = b2[0], s1 = b2[1];
  #pragma unroll
  for (int c = 0; c < 32; ++c) { s0 += W2[c] * h[c]; s1 += W2[32 + c] * h[c]; }
  score[(long long)bb * 2 * NQ + n]      = s0;
  score[(long long)bb * 2 * NQ + NQ + n] = s1;
  mask[(long long)bb * NQ + n] = (s1 > s0) ? 1 : 0;   // argmax over channel dim
}

// ---------------- qidx output (padded index = tgt index + 1) ----------------
__global__ void qidx_kernel(const int* __restrict__ idxbuf, const int* __restrict__ cntbuf,
                            const int* __restrict__ mask, float* __restrict__ out, int NQ) {
  long long t = (long long)blockIdx.x * blockDim.x + threadIdx.x;
  int bb = blockIdx.z;
  if (t >= (long long)NQ * kNS) return;
  int s = (int)(t % kNS);
  int q = (int)(t / kNS);
  int cnt = cntbuf[(long long)bb * NQ + q];
  int m   = mask[(long long)bb * NQ + q];
  int v = (cnt > 0 && m) ? idxbuf[((long long)bb * NQ + q) * kNS + s] + 1 : 0;
  out[(long long)bb * NQ * kNS + t] = (float)v;
}

// ---------------- masked transpose (B,32,P,J) -> (B, 32*BINS, RAY*P) ----------------
__global__ void bintrans_kernel(const float* __restrict__ feat, const int* __restrict__ mask,
                                float* __restrict__ dst, int NQ, int BINS) {
  long long t = (long long)blockIdx.x * blockDim.x + threadIdx.x;
  int bb = blockIdx.z;
  if (t >= (long long)kHALF * NQ) return;
  int q = (int)(t % NQ);
  int h = (int)(t / NQ);
  int JP  = BINS * kRAY;
  int p   = q / JP, j = q % JP;
  int bin = j / kRAY, ray = j % kRAY;
  float v = feat[(long long)bb * kHALF * NQ + t] * (float)mask[(long long)bb * NQ + q];
  dst[(long long)bb * (kHALF * BINS) * (kRAY * kP) +
      (long long)(h * BINS + bin) * (kRAY * kP) + ray * kP + p] = v;
}

// =====================================================================
extern "C" void kernel_launch(void* const* d_in, const int* in_sizes, int n_in,
                              void* d_out, int out_size, void* d_ws, size_t ws_size,
                              hipStream_t stream) {
  (void)in_sizes; (void)n_in; (void)out_size; (void)ws_size;
  // ---- inputs (setup_inputs dict order; params assumed in insertion order) ----
  const float* seed_xyz   = (const float*)d_in[0];
  const float* seed_feat  = (const float*)d_in[1];
  /* d_in[2] proposals: shape only */
  const float* scale_pred = (const float*)d_in[3];
  const float* ref_points = (const float*)d_in[4];
  const float* points_cat = (const float*)d_in[5];
  const float* aggf       = (const float*)d_in[6];
  int pi = 7;
  const float* sr_W1 = (const float*)d_in[pi++]; const float* sr_b1 = (const float*)d_in[pi++];
  const float* sr_W2 = (const float*)d_in[pi++]; const float* sr_b2 = (const float*)d_in[pi++];
  const float* cs_W  = (const float*)d_in[pi++]; const float* cs_b  = (const float*)d_in[pi++];
  const float* fs_W  = (const float*)d_in[pi++]; const float* fs_b  = (const float*)d_in[pi++];
  const float* ci_W1 = (const float*)d_in[pi++]; const float* ci_b1 = (const float*)d_in[pi++];
  const float* ci_W2 = (const float*)d_in[pi++]; const float* ci_b2 = (const float*)d_in[pi++];
  const float* fi_W1 = (const float*)d_in[pi++]; const float* fi_b1 = (const float*)d_in[pi++];
  const float* fi_W2 = (const float*)d_in[pi++]; const float* fi_b2 = (const float*)d_in[pi++];
  const float* cb_W  = (const float*)d_in[pi++]; const float* cb_b  = (const float*)d_in[pi++];
  const float* fb_W  = (const float*)d_in[pi++]; const float* fb_b  = (const float*)d_in[pi++];
  const float* cr_W1 = (const float*)d_in[pi++]; const float* cr_b1 = (const float*)d_in[pi++];
  const float* cr_W2 = (const float*)d_in[pi++]; const float* cr_b2 = (const float*)d_in[pi++];
  const float* fr_W1 = (const float*)d_in[pi++]; const float* fr_b1 = (const float*)d_in[pi++];
  const float* fr_W2 = (const float*)d_in[pi++]; const float* fr_b2 = (const float*)d_in[pi++];
  const float* fu_W1 = (const float*)d_in[pi++]; const float* fu_b1 = (const float*)d_in[pi++];
  const float* fu_W2 = (const float*)d_in[pi++]; const float* fu_b2 = (const float*)d_in[pi++];

  // ---- workspace layout ----
  size_t off = 0;
  auto take = [&](size_t bytes) -> char* {
    char* r = (char*)d_ws + off;
    off = (off + bytes + 255) & ~(size_t)255;
    return r;
  };
  float* rays     = (float*)take(kRAY * 3 * 4);
  float* tgt      = (float*)take((size_t)kB * kFPSM * 3 * 4);
  int*   nn_idx   = (int*)  take((size_t)kB * kFPSM * 3 * 4);
  float* nn_w     = (float*)take((size_t)kB * kFPSM * 3 * 4);
  float* i256     = (float*)take((size_t)kB * kSEED * kFPSM * 4);
  float* i128     = (float*)take((size_t)kB * 128 * kFPSM * 4);
  float* i64      = (float*)take((size_t)kB * kRED * kFPSM * 4);
  float* c_pos    = (float*)take((size_t)kB * kNQC * 3 * 4);
  float* f_pos    = (float*)take((size_t)kB * kNQF * 3 * 4);
  int*   c_idx    = (int*)  take((size_t)kB * kNQC * kNS * 4);
  int*   c_cnt    = (int*)  take((size_t)kB * kNQC * 4);
  int*   f_idx    = (int*)  take((size_t)kB * kNQF * kNS * 4);
  int*   f_cnt    = (int*)  take((size_t)kB * kNQF * 4);
  float* c_feat   = (float*)take((size_t)kB * kHALF * kNQC * 4);
  float* f_feat   = (float*)take((size_t)kB * kHALF * kNQF * 4);
  float* ha_c     = (float*)take((size_t)kB * 32 * kP * 4);
  float* ha_f     = (float*)take((size_t)kB * 32 * kP * 4);
  float* gb_c     = (float*)take((size_t)kB * 32 * kNQC * 4);
  float* gb_f     = (float*)take((size_t)kB * 32 * kNQF * 4);
  int*   c_mask   = (int*)  take((size_t)kB * kNQC * 4);
  int*   f_mask   = (int*)  take((size_t)kB * kNQF * 4);
  float* dmaxb    = (float*)take((size_t)kB * kP * kRAY * 4);
  float* cf_in    = (float*)take((size_t)kB * 128 * (kRAY * kP) * 4);
  float* ff_in    = (float*)take((size_t)kB * 64 * (kRAY * kP) * 4);
  float* cbin_o   = (float*)take((size_t)kB * 32 * (kRAY * kP) * 4);
  float* fbin_o   = (float*)take((size_t)kB * 32 * (kRAY * kP) * 4);
  float* rayh     = (float*)take((size_t)kB * 256 * kP * 4);
  float* fuse_in  = (float*)take((size_t)kB * 256 * kP * 4);
  float* fuse_h   = (float*)take((size_t)kB * 256 * kP * 4);

  // ---- output layout (return order: fuse, f_score, f_qidx, c_score, c_qidx) ----
  float* out        = (float*)d_out;
  float* out_fuse   = out;
  float* out_fscore = out + (size_t)kB * kAGG * kP;                  // 32768
  float* out_fqidx  = out_fscore + (size_t)kB * 2 * kNQF;            // +67584
  float* out_cscore = out_fqidx + (size_t)kB * kNQF * kNS;           // +540672
  float* out_cqidx  = out_cscore + (size_t)kB * 2 * kNQC;            // +135168

  auto gemm = [&](const float* A, int lda, const float* Bm, long long sB,
                  const float* bias, float* C, long long sC,
                  int M, int N, int K, int relu) {
    dim3 g((N + 15) / 16, M / 16, kB);
    gemm16_wmma<<<g, dim3(32), 0, stream>>>(A, lda, Bm, sB, bias, C, sC, N, K, relu);
  };
  auto ew = [&](int per_batch) {
    return dim3((per_batch + 255) / 256, 1, kB);
  };

  // 1) ray table + FPS (whole cloud in LDS: 3*16384 + 3*1024 dwords = 208896 B)
  rays_kernel<<<1, 1, 0, stream>>>(rays);
  fps_kernel<<<kB, 1024, (3 * kNPTS + 3 * 1024) * sizeof(float), stream>>>(points_cat, tgt);

  // 2) three_nn + feature interpolation + seed_feat_reduce MLP (WMMA)
  nn_kernel<<<ew(kFPSM), 256, 0, stream>>>(tgt, seed_xyz, nn_idx, nn_w);
  interp_gather_kernel<<<ew(kSEED * kFPSM), 256, 0, stream>>>(seed_feat, nn_idx, nn_w, i256);
  gemm(sr_W1, 256, i256, (long long)kSEED * kFPSM, sr_b1, i128, (long long)128 * kFPSM,
       128, kFPSM, 256, 1);
  gemm(sr_W2, 128, i128, (long long)128 * kFPSM, sr_b2, i64, (long long)kRED * kFPSM,
       64, kFPSM, 128, 0);

  // 3) coarse branch
  cpos_kernel<<<ew(kNQC), 256, 0, stream>>>(rays, scale_pred, ref_points, c_pos);
  ballq_kernel<<<dim3(kNQC, 1, kB), 32, 0, stream>>>(tgt, c_pos, c_idx, c_cnt, kSAR2, kNQC);
  sa_wmma<<<dim3(kNQC / 8, 1, kB), 32, 0, stream>>>(cs_W, cs_b, i64, c_idx, c_cnt, c_feat, kNQC);
  // coarse_inter split: W1[:, :128] @ agg (per-proposal) + W1[:, 128:] @ c_feat
  gemm(ci_W1, 160, aggf, (long long)kAGG * kP, nullptr, ha_c, (long long)32 * kP,
       32, kP, 128, 0);
  gemm(ci_W1 + 128, 160, c_feat, (long long)kHALF * kNQC, nullptr, gb_c, (long long)32 * kNQC,
       32, kNQC, 32, 0);
  inter2_kernel<<<ew(kNQC), 256, 0, stream>>>(gb_c, ha_c, ci_b1, ci_W2, ci_b2,
                                              out_cscore, c_mask, kNQC, kBIN * kRAY);
  qidx_kernel<<<ew(kNQC * kNS), 256, 0, stream>>>(c_idx, c_cnt, c_mask, out_cqidx, kNQC);

  // 4) fine branch (radii derived from masked coarse targets)
  dmax_kernel<<<ew(kP * kRAY), 256, 0, stream>>>(scale_pred, c_mask, dmaxb);
  fpos_kernel<<<ew(kNQF), 256, 0, stream>>>(rays, dmaxb, ref_points, f_pos);
  ballq_kernel<<<dim3(kNQF, 1, kB), 32, 0, stream>>>(tgt, f_pos, f_idx, f_cnt, kFSAR2, kNQF);
  sa_wmma<<<dim3(kNQF / 8, 1, kB), 32, 0, stream>>>(fs_W, fs_b, i64, f_idx, f_cnt, f_feat, kNQF);
  gemm(fi_W1, 160, aggf, (long long)kAGG * kP, nullptr, ha_f, (long long)32 * kP,
       32, kP, 128, 0);
  gemm(fi_W1 + 128, 160, f_feat, (long long)kHALF * kNQF, nullptr, gb_f, (long long)32 * kNQF,
       32, kNQF, 32, 0);
  inter2_kernel<<<ew(kNQF), 256, 0, stream>>>(gb_f, ha_f, fi_b1, fi_W2, fi_b2,
                                              out_fscore, f_mask, kNQF, kFBIN * kRAY);
  qidx_kernel<<<ew(kNQF * kNS), 256, 0, stream>>>(f_idx, f_cnt, f_mask, out_fqidx, kNQF);

  // 5) bin + ray MLP chains -> fuse_in ([f_out ; c_out])
  const int RP = kRAY * kP;  // 8448
  bintrans_kernel<<<ew(kHALF * kNQC), 256, 0, stream>>>(c_feat, c_mask, cf_in, kNQC, kBIN);
  gemm(cb_W, 128, cf_in, (long long)128 * RP, cb_b, cbin_o, (long long)32 * RP, 32, RP, 128, 0);
  gemm(cr_W1, kHALF * kRAY, cbin_o, (long long)32 * RP, cr_b1, rayh, (long long)256 * kP,
       256, kP, kHALF * kRAY, 1);
  gemm(cr_W2, 256, rayh, (long long)256 * kP, cr_b2, fuse_in + (size_t)kAGG * kP,
       (long long)256 * kP, 128, kP, 256, 0);                      // c_out -> rows 128..255
  bintrans_kernel<<<ew(kHALF * kNQF), 256, 0, stream>>>(f_feat, f_mask, ff_in, kNQF, kFBIN);
  gemm(fb_W, 64, ff_in, (long long)64 * RP, fb_b, fbin_o, (long long)32 * RP, 32, RP, 64, 0);
  gemm(fr_W1, kHALF * kRAY, fbin_o, (long long)32 * RP, fr_b1, rayh, (long long)256 * kP,
       256, kP, kHALF * kRAY, 1);
  gemm(fr_W2, 256, rayh, (long long)256 * kP, fr_b2, fuse_in,
       (long long)256 * kP, 128, kP, 256, 0);                      // f_out -> rows 0..127

  // 6) fuse MLP -> output
  gemm(fu_W1, 256, fuse_in, (long long)256 * kP, fu_b1, fuse_h, (long long)256 * kP,
       256, kP, 256, 1);
  gemm(fu_W2, 256, fuse_h, (long long)256 * kP, fu_b2, out_fuse, (long long)kAGG * kP,
       128, kP, 256, 0);
}
